// CAModel3D_68315749810834
// MI455X (gfx1250) — matrix operand order repaired and checked by reading, not compiled, not run on hardware
//
#include <hip/hip_runtime.h>
#include <cstdint>

typedef __attribute__((ext_vector_type(16))) _Float16 v16h;
typedef __attribute__((ext_vector_type(8)))  _Float16 v8h;
typedef __attribute__((ext_vector_type(8)))  float    v8f;
typedef __attribute__((ext_vector_type(4)))  float    f32x4;
typedef __attribute__((ext_vector_type(4)))  unsigned int u32x4;
typedef __attribute__((ext_vector_type(8)))  int      i32x8;
typedef __attribute__((ext_vector_type(4)))  int      i32x4;

#define NB 4
#define DD 64
#define CH 16

// ---------------------------------------------------------------------------
// CDNA5 async / TDM helpers (signatures per round-2 compiler feedback)
// ---------------------------------------------------------------------------
__device__ __forceinline__ void async_copy_b32(const float* g, float* l) {
#if __has_builtin(__builtin_amdgcn_global_load_async_to_lds_b32)
  __builtin_amdgcn_global_load_async_to_lds_b32(
      (__attribute__((address_space(1))) int*)g,
      (__attribute__((address_space(3))) int*)l, 0, 0);
#else
  asm volatile("global_load_async_to_lds_b32 %0, %1, off"
               :: "v"((unsigned int)(uintptr_t)l), "v"(g)
               : "memory");
#endif
}

__device__ __forceinline__ void wait_async0() {
#if __has_builtin(__builtin_amdgcn_s_wait_asynccnt)
  __builtin_amdgcn_s_wait_asynccnt(0);
#else
  asm volatile("s_wait_asynccnt 0x0" ::: "memory");
#endif
}

#if __has_builtin(__builtin_amdgcn_tensor_load_to_lds)
#define HAVE_TDM 1
// 2-D tensor DMA: global (height x width f32, row-major, stride = width) -> LDS.
// D# packing per CDNA5 ISA. 6-arg builtin form.
__device__ __forceinline__ void tdm_load_2d(unsigned int lds_addr,
                                            unsigned long long gaddr,
                                            unsigned int width,
                                            unsigned int height) {
  u32x4 g0;
  g0.x = 1u;                                        // count=1, no gather
  g0.y = lds_addr;                                  // lds_addr [63:32]
  g0.z = (unsigned int)gaddr;                       // global_addr lo
  g0.w = (unsigned int)(gaddr >> 32) | (2u << 30);  // global_addr hi | type=2
  i32x8 g1;
  g1[0] = (int)(2u << 16);           // workgroup_mask=0, data_size=2 (4B)
  g1[1] = (int)(width << 16);        // atomic_barrier=0 | tensor_dim0[15:0]
  g1[2] = (int)(height << 16);       // tensor_dim0 hi | tensor_dim1[15:0]
  g1[3] = (int)(width << 16);        // tensor_dim1 hi | tile_dim0
  g1[4] = (int)(height & 0xffffu);   // tile_dim1 | tile_dim2=0
  g1[5] = (int)width;                // tensor_dim0_stride lo32
  g1[6] = 0;                         // stride0 hi | tensor_dim1_stride lo
  g1[7] = 0;
  i32x4 z4 = (i32x4)0;               // groups 2/3 unused (2-D tensor)
  i32x8 z8 = (i32x8)0;               // 5th arg (6-arg toolchain variant)
  __builtin_amdgcn_tensor_load_to_lds(g0, g1, z4, z4, z8, 0);
}
#endif

// ---------------------------------------------------------------------------
// Kernel 1: fused perception (depthwise 3D stencil) + 2-layer MLP via WMMA +
// stochastic update. One wave = 16 voxels (M=16). Block = 256 thr = 8 waves
// = 128 consecutive voxels (2 x-rows). Writes x_new to workspace.
// ---------------------------------------------------------------------------
__global__ __launch_bounds__(256) void nca_update_kernel(
    const float* __restrict__ x,
    const float* __restrict__ w1, const float* __restrict__ gb1,
    const float* __restrict__ w2, const float* __restrict__ gb2,
    const float* __restrict__ rand_u,
    float* __restrict__ xnew)
{
  __shared__ _Float16 w1t[128 * 64];     // w1 transposed [n][k] f16 (16KB)
  __shared__ _Float16 w2t[16 * 128];     // w2 transposed [n][k] f16 (4KB)
  __shared__ _Float16 hbuf[8][16 * 128]; // per-wave hidden acts (32KB)

  const int tid  = threadIdx.x;
  const int wave = tid >> 5;
  const int lane = tid & 31;
  const int m    = lane & 15;   // A row / B,C,D column owned by lane
  const int half = lane >> 4;   // lane group (selects K sub-ranges)

  // ---- stage weights to LDS as f16, transposed ([n][k]) ----
#if defined(HAVE_TDM)
  __shared__ float w1raw[64 * 128];      // 32KB raw f32 staging via TDM
  __shared__ float w2raw[128 * 16];      // 8KB
  if (wave == 0) {                       // one wave issues DMA descriptors
    tdm_load_2d((unsigned int)(uintptr_t)w1raw,
                (unsigned long long)(uintptr_t)w1, 128u, 64u);
    tdm_load_2d((unsigned int)(uintptr_t)w2raw,
                (unsigned long long)(uintptr_t)w2, 16u, 128u);
    __builtin_amdgcn_s_wait_tensorcnt(0);
  }
  __syncthreads();
  for (int i = tid; i < 64 * 128; i += 256) {
    const int k = i >> 7, n = i & 127;
    w1t[n * 64 + k] = (_Float16)w1raw[i];
  }
  for (int i = tid; i < 128 * 16; i += 256) {
    const int k = i >> 4, n = i & 15;
    w2t[n * 128 + k] = (_Float16)w2raw[i];
  }
#else
  for (int i = tid; i < 64 * 128; i += 256) {
    const int k = i >> 7, n = i & 127;          // w1 is [64][128] row-major
    w1t[n * 64 + k] = (_Float16)w1[i];
  }
  for (int i = tid; i < 128 * 16; i += 256) {
    const int k = i >> 4, n = i & 15;           // w2 is [128][16] row-major
    w2t[n * 128 + k] = (_Float16)w2[i];
  }
#endif
  __syncthreads();

  // ---- block -> (batch, z, y-pair); wave voxel = 16*wave + m ----
  const int blk = blockIdx.x;              // 4*64*32 blocks
  const int y0  = (blk & 31) * 2;
  const int z   = (blk >> 5) & 63;
  const int b   = blk >> 11;

  const int vloc = wave * 16 + m;          // 0..127 within block
  const int xx   = vloc & 63;
  const int yy   = y0 + (vloc >> 6);

  {
    const size_t vc = (((size_t)(b * 64 + z) * 64 + yy) * 64 + xx) * CH;
    __builtin_prefetch(x + vc, 0, 0);      // global_prefetch_b8
  }

  // ---- perception: build layer-1 A fragments (16x32 f16, two K-chunks) ----
  // Lane owns channels c = kt*8 + half*4 + {0,1,2,3}; each gives 4 features
  // (identity, d/dz, d/dy, d/dx Sobel) at fragment slots {0,4,8,12}+f.
  const float S[3] = {1.f, 2.f, 1.f};
  const float G[3] = {1.f, 0.f, -1.f};
  v16h afr[2];
  #pragma unroll
  for (int kt = 0; kt < 2; ++kt) {
    #pragma unroll
    for (int p = 0; p < 2; ++p) {
      const int c0 = kt * 8 + half * 4 + p * 2;   // channel pair c0, c0+1
      float id0 = 0.f, id1 = 0.f;
      float s1a = 0.f, s2a = 0.f, s3a = 0.f;
      float s1b = 0.f, s2b = 0.f, s3b = 0.f;
      #pragma unroll
      for (int dz = -1; dz <= 1; ++dz) {
        const int zz = z + dz; if ((unsigned)zz >= 64u) continue;
        #pragma unroll
        for (int dy = -1; dy <= 1; ++dy) {
          const int yb = yy + dy; if ((unsigned)yb >= 64u) continue;
          #pragma unroll
          for (int dxv = -1; dxv <= 1; ++dxv) {
            const int xb = xx + dxv; if ((unsigned)xb >= 64u) continue;
            const float* pn =
                x + ((((size_t)(b * 64 + zz) * 64 + yb) * 64 + xb) * CH + c0);
            const float v0 = pn[0], v1 = pn[1];   // merged to b64 load
            const float f1 = G[dz + 1] * S[dy + 1] * S[dxv + 1];
            const float f2 = S[dz + 1] * G[dy + 1] * S[dxv + 1];
            const float f3 = S[dz + 1] * S[dy + 1] * G[dxv + 1];
            s1a = fmaf(f1, v0, s1a); s2a = fmaf(f2, v0, s2a); s3a = fmaf(f3, v0, s3a);
            s1b = fmaf(f1, v1, s1b); s2b = fmaf(f2, v1, s2b); s3b = fmaf(f3, v1, s3b);
            if (dz == 0 && dy == 0 && dxv == 0) { id0 = v0; id1 = v1; }
          }
        }
      }
      const float inv32 = 0.03125f;
      afr[kt][p * 8 + 0] = (_Float16)id0;
      afr[kt][p * 8 + 1] = (_Float16)(s1a * inv32);
      afr[kt][p * 8 + 2] = (_Float16)(s2a * inv32);
      afr[kt][p * 8 + 3] = (_Float16)(s3a * inv32);
      afr[kt][p * 8 + 4] = (_Float16)id1;
      afr[kt][p * 8 + 5] = (_Float16)(s1b * inv32);
      afr[kt][p * 8 + 6] = (_Float16)(s2b * inv32);
      afr[kt][p * 8 + 7] = (_Float16)(s3b * inv32);
    }
  }

  // ---- layer 1: (16x64) @ (64x128) as 8 N-tiles x 2 K-chunks of WMMA ----
  const int nB = m;
  #pragma unroll
  for (int nt = 0; nt < 8; ++nt) {
    const v16h bf0 = *(const v16h*)&w1t[(nt * 16 + nB) * 64 + half * 16];
    const v16h bf1 = *(const v16h*)&w1t[(nt * 16 + nB) * 64 + 32 + half * 16];
    v8f acc = {};
    acc = __builtin_amdgcn_wmma_f32_16x16x32_f16(false, afr[0], false, bf0,
                                                 (short)0, acc, false, false);
    acc = __builtin_amdgcn_wmma_f32_16x16x32_f16(false, afr[1], false, bf1,
                                                 (short)0, acc, false, false);
    const float bias = gb1[nt * 16 + nB];
    #pragma unroll
    for (int r = 0; r < 8; ++r) {          // D row M = r + 8*half
      float hv = acc[r] + bias;
      hv = hv > 0.f ? hv : 0.f;            // relu
      hbuf[wave][(r + 8 * half) * 128 + nt * 16 + nB] = (_Float16)hv;
    }
  }

  // ---- layer 2: (16x128) @ (128x16), 4 chained K=32 WMMAs ----
  v8f acc2 = {};
  #pragma unroll
  for (int kt2 = 0; kt2 < 4; ++kt2) {
    const _Float16* hrow = &hbuf[wave][m * 128 + kt2 * 32];
    const v8h lo = *(const v8h*)(hrow + half * 16);       // K = half*16 + j
    const v8h hi = *(const v8h*)(hrow + half * 16 + 8);   // K = half*16+8+j
    const v16h a2 = __builtin_shufflevector(lo, hi, 0, 1, 2, 3, 4, 5, 6, 7,
                                            8, 9, 10, 11, 12, 13, 14, 15);
    const v16h b2f = *(const v16h*)&w2t[nB * 128 + kt2 * 32 + half * 16];
    acc2 = __builtin_amdgcn_wmma_f32_16x16x32_f16(false, a2, false, b2f,
                                                  (short)0, acc2, false, false);
  }

  // ---- stochastic update: x_new = x + dx * (rand <= 0.5) ----
  const float bias2 = gb2[nB];
  #pragma unroll
  for (int r = 0; r < 8; ++r) {
    const int vm = wave * 16 + r + 8 * half;   // voxel within block
    const int gx = vm & 63;
    const int gy = y0 + (vm >> 6);
    const size_t vidx = ((size_t)(b * 64 + z) * 64 + gy) * 64 + gx;
    const float ru  = rand_u[vidx];
    const float mk  = (ru <= 0.5f) ? 1.f : 0.f;
    const float dxv = acc2[r] + bias2;         // STEP_SIZE == 1
    xnew[vidx * CH + nB] = x[vidx * CH + nB] + dxv * mk;
  }
}

// ---------------------------------------------------------------------------
// Kernel 2: life mask. Alpha slabs (old + new) are gathered into LDS with
// async-to-LDS loads (one 4B element per lane), then both 3x3x3 max-pools
// read LDS. Block = (1 z) x (4 y) x (64 x) = 256 voxels; halo staged with
// -inf padding (matches reduce_window init). Output stores are nontemporal
// (write-once; keep L2 for the x/x_new stencil reuse).
// ---------------------------------------------------------------------------
__global__ __launch_bounds__(256) void nca_life_kernel(
    const float* __restrict__ xold,
    const float* __restrict__ xnew,
    float* __restrict__ out)
{
  __shared__ float aold[3 * 6 * 64];   // [z-1..z+1][y0-1..y0+4][x]
  __shared__ float anew[3 * 6 * 64];

  const int blk = blockIdx.x;          // 4 * 64 * 16 blocks
  const int y0  = (blk & 15) * 4;
  const int z   = (blk >> 4) & 63;
  const int b   = blk >> 10;

  for (int i = threadIdx.x; i < 3 * 6 * 64; i += 256) {
    const int xs = i & 63;
    const int ys = (i >> 6) % 6;
    const int zs = i / (6 * 64);
    const int gy = y0 - 1 + ys;
    const int gz = z - 1 + zs;
    if ((unsigned)gy >= 64u || (unsigned)gz >= 64u) {
      aold[i] = -3.4e38f;              // OOB halo = -inf (reduce_window init)
      anew[i] = -3.4e38f;
    } else {
      const size_t nb = (((size_t)(b * 64 + gz) * 64 + gy) * 64 + xs) * CH + 3;
      async_copy_b32(xold + nb, &aold[i]);
      async_copy_b32(xnew + nb, &anew[i]);
    }
  }
  wait_async0();
  __syncthreads();

  const int ly = threadIdx.x >> 6;     // 0..3
  const int xx = threadIdx.x & 63;
  float mo = -3.4e38f, mn = -3.4e38f;
  #pragma unroll
  for (int dz = 0; dz < 3; ++dz) {
    #pragma unroll
    for (int dy = 0; dy < 3; ++dy) {
      const int base = (dz * 6 + (ly + dy)) * 64;
      #pragma unroll
      for (int dx = -1; dx <= 1; ++dx) {
        const int xb = xx + dx;
        if ((unsigned)xb >= 64u) continue;
        mo = fmaxf(mo, aold[base + xb]);
        mn = fmaxf(mn, anew[base + xb]);
      }
    }
  }
  const float life = (mo > 0.1f && mn > 0.1f) ? 1.f : 0.f;

  const size_t v = ((size_t)(b * 64 + z) * 64 + (y0 + ly)) * 64 + xx;
  const f32x4* src = (const f32x4*)(xnew + v * CH);
  f32x4*       dst = (f32x4*)(out + v * CH);
  #pragma unroll
  for (int i = 0; i < 4; ++i) {
    f32x4 t = src[i];
    t *= life;
    __builtin_nontemporal_store(t, &dst[i]);
  }
}

// ---------------------------------------------------------------------------
extern "C" void kernel_launch(void* const* d_in, const int* in_sizes, int n_in,
                              void* d_out, int out_size, void* d_ws, size_t ws_size,
                              hipStream_t stream) {
  const float* x   = (const float*)d_in[0];
  const float* w1  = (const float*)d_in[1];
  const float* b1  = (const float*)d_in[2];
  const float* w2  = (const float*)d_in[3];
  const float* b2  = (const float*)d_in[4];
  const float* ru  = (const float*)d_in[5];
  float* out  = (float*)d_out;
  float* xnew = (float*)d_ws;              // needs 4*64^3*16*4 = 64 MiB

  // Kernel 1: 4 batches * 64 z * 32 y-pairs = 8192 blocks of 256 threads.
  nca_update_kernel<<<NB * DD * (DD / 2), 256, 0, stream>>>(
      x, w1, b1, w2, b2, ru, xnew);
  // Kernel 2: 4 batches * 64 z * 16 y-quads = 4096 blocks, 1 voxel/thread.
  nca_life_kernel<<<NB * DD * (DD / 4), 256, 0, stream>>>(
      x, xnew, out);
}
